// Model_33208687132773
// MI455X (gfx1250) — compile-verified, matrix operand order checked by reading, not run on hardware
//
#include <hip/hip_runtime.h>
#include <hip/hip_bf16.h>
#include <math.h>

typedef __attribute__((ext_vector_type(16))) _Float16 v16h;
typedef __attribute__((ext_vector_type(8)))  float    v8f;

// ---------------- model constants ----------------
namespace {
constexpr int NB   = 32;            // batch
constexpr int SEQ  = 336;
constexpr int LBL  = 168;
constexpr int PRED = 336;
constexpr int CIN  = 321;
constexpr int DM   = 512;
constexpr int NH   = 8;
constexpr int DFF  = 2048;
constexpr int EH   = 64;            // head dim
constexpr int COUT = 321;
constexpr int LDEC = LBL + PRED;    // 504
constexpr int LQE  = 336;           // enc queries (mult of 16)
constexpr int LKE  = 352;           // enc keys padded to mult 32
constexpr int LQD  = 512;           // dec queries padded
constexpr int LKD  = 512;           // dec keys padded
constexpr int K3   = 992;           // 3*CIN=963 padded to mult 32
constexpr int KTR  = 352;           // SEQ padded for trend MLP
}

// ================= WMMA GEMM: C[M,N] = A[M,K]f16 * W[N,K]f16^T (+bias)(+relu) ===========
// K must be a multiple of 32 (callers zero-pad). 256 threads = 8 waves, tile 128x128,
// each wave computes 64x32 via a 4x2 grid of v_wmma_f32_16x16x32_f16.
__global__ __launch_bounds__(256)
void gemm_f16_kernel(const _Float16* __restrict__ A, int lda,
                     const _Float16* __restrict__ W, int ldw,
                     float* __restrict__ C, int ldc,
                     const float* __restrict__ bias,
                     int M, int N, int K, int relu)
{
    __shared__ __align__(32) _Float16 As[128 * 32];
    __shared__ __align__(32) _Float16 Ws[128 * 32];
    const int tid  = threadIdx.x;
    const int lane = tid & 31, wid = tid >> 5;
    const int wm = wid & 1, wn = wid >> 1;           // 2 x 4 wave grid
    const int m0 = blockIdx.x * 128, n0 = blockIdx.y * 128;
    const int ln15 = lane & 15, hi = lane >> 4;
    const int lrow = tid >> 1, lcol = (tid & 1) * 16; // 32B per thread per tile

    v8f acc[4][2];
#pragma unroll
    for (int mt = 0; mt < 4; ++mt)
#pragma unroll
        for (int nt = 0; nt < 2; ++nt) acc[mt][nt] = (v8f){};

    for (int k0 = 0; k0 < K; k0 += 32) {
        {   // stage A and W slabs into LDS (zero OOB rows)
            v16h va = {};
            const int gm = m0 + lrow;
            if (gm < M) va = *(const v16h*)(A + (size_t)gm * lda + k0 + lcol);
            *(v16h*)(As + lrow * 32 + lcol) = va;
            v16h vw = {};
            const int gn = n0 + lrow;
            if (gn < N) vw = *(const v16h*)(W + (size_t)gn * ldw + k0 + lcol);
            *(v16h*)(Ws + lrow * 32 + lcol) = vw;
        }
        __syncthreads();
        if (k0 + 32 < K) {  // software prefetch of next K slab -> global_prefetch_b8
            const int gm = m0 + lrow;
            if (gm < M) __builtin_prefetch(A + (size_t)gm * lda + k0 + 32 + lcol, 0, 0);
            const int gn = n0 + lrow;
            if (gn < N) __builtin_prefetch(W + (size_t)gn * ldw + k0 + 32 + lcol, 0, 0);
        }
        v16h bfr[2];
#pragma unroll
        for (int nt = 0; nt < 2; ++nt)
            bfr[nt] = *(const v16h*)(Ws + (wn * 32 + nt * 16 + ln15) * 32 + hi * 16);
#pragma unroll
        for (int mt = 0; mt < 4; ++mt) {
            v16h afr = *(const v16h*)(As + (wm * 64 + mt * 16 + ln15) * 32 + hi * 16);
#pragma unroll
            for (int nt = 0; nt < 2; ++nt)
                acc[mt][nt] = __builtin_amdgcn_wmma_f32_16x16x32_f16(
                    false, afr, false, bfr[nt], (short)0, acc[mt][nt], false, false);
        }
        __syncthreads();
    }
    // epilogue: C layout vgpr r -> (M=r+8*hi, N=ln15)
#pragma unroll
    for (int mt = 0; mt < 4; ++mt)
#pragma unroll
        for (int nt = 0; nt < 2; ++nt) {
            const int gn = n0 + wn * 32 + nt * 16 + ln15;
            const float bv = (bias && gn < N) ? bias[gn] : 0.f;
#pragma unroll
            for (int r = 0; r < 8; ++r) {
                const int gm = m0 + wm * 64 + mt * 16 + r + 8 * hi;
                if (gm < M && gn < N) {
                    float v = acc[mt][nt][r] + bv;
                    if (relu) v = fmaxf(v, 0.f);
                    C[(size_t)gm * ldc + gn] = v;
                }
            }
        }
}

// ============ Fused flash attention (de-stationary): O = softmax((QK^T*tau+delta)*scale) V =====
// 1 wave / 16 queries; 32-key blocks; Q,K in [B,H,Lpad,E] f16; V pre-transposed [B,H,E,Lkpad] f16.
__global__ __launch_bounds__(32)
void attn_kernel(const _Float16* __restrict__ Qm, const _Float16* __restrict__ Km,
                 const _Float16* __restrict__ Vt, float* __restrict__ Om,
                 const float* __restrict__ tau, const float* __restrict__ delta,
                 int Lq, int Lk, int Lqpad, int Lkpad, int causal, float scale)
{
    __shared__ __align__(32) _Float16 Pl[16 * 32];
    const int lane = threadIdx.x;
    const int ln15 = lane & 15, hi = lane >> 4;
    const int bh = blockIdx.y;
    const int b = bh / NH, h = bh % NH;
    const int q0 = blockIdx.x * 16;

    const _Float16* Qb = Qm + (size_t)bh * Lqpad * EH;
    const _Float16* Kb = Km + (size_t)bh * Lkpad * EH;
    const _Float16* Vb = Vt + (size_t)bh * EH * Lkpad;

    v16h aq0 = *(const v16h*)(Qb + (size_t)(q0 + ln15) * EH + hi * 16);
    v16h aq1 = *(const v16h*)(Qb + (size_t)(q0 + ln15) * EH + 32 + hi * 16);

    v8f acc0 = {}, acc1 = {}, acc2 = {}, acc3 = {};
    float mrun[8], lrun[8];
#pragma unroll
    for (int r = 0; r < 8; ++r) { mrun[r] = -1e30f; lrun[r] = 0.f; }
    const float tv = tau[b];

    for (int j = 0; j < Lkpad; j += 32) {
        v16h bk0a = *(const v16h*)(Kb + (size_t)(j + ln15) * EH + hi * 16);
        v16h bk0b = *(const v16h*)(Kb + (size_t)(j + ln15) * EH + 32 + hi * 16);
        v16h bk1a = *(const v16h*)(Kb + (size_t)(j + 16 + ln15) * EH + hi * 16);
        v16h bk1b = *(const v16h*)(Kb + (size_t)(j + 16 + ln15) * EH + 32 + hi * 16);
        v8f s0 = {}, s1 = {};
        s0 = __builtin_amdgcn_wmma_f32_16x16x32_f16(false, aq0, false, bk0a, (short)0, s0, false, false);
        s0 = __builtin_amdgcn_wmma_f32_16x16x32_f16(false, aq1, false, bk0b, (short)0, s0, false, false);
        s1 = __builtin_amdgcn_wmma_f32_16x16x32_f16(false, aq0, false, bk1a, (short)0, s1, false, false);
        s1 = __builtin_amdgcn_wmma_f32_16x16x32_f16(false, aq1, false, bk1b, (short)0, s1, false, false);

        const int k0i = j + ln15;
        const int k1i = j + 16 + ln15;
        float d0 = 0.f, d1 = 0.f;
        if (delta) {
            d0 = delta[(size_t)b * Lk + (k0i < Lk ? k0i : Lk - 1)];
            d1 = delta[(size_t)b * Lk + (k1i < Lk ? k1i : Lk - 1)];
        }
        float p0[8], p1[8], alpha[8];
#pragma unroll
        for (int r = 0; r < 8; ++r) {
            const int qi = q0 + r + 8 * hi;
            float v0 = (s0[r] * tv + d0) * scale;
            float v1 = (s1[r] * tv + d1) * scale;
            if (k0i >= Lk || (causal && k0i > qi)) v0 = -1e30f;
            if (k1i >= Lk || (causal && k1i > qi)) v1 = -1e30f;
            float mx = fmaxf(v0, v1);
            mx = fmaxf(mx, __shfl_xor(mx, 1, 32));
            mx = fmaxf(mx, __shfl_xor(mx, 2, 32));
            mx = fmaxf(mx, __shfl_xor(mx, 4, 32));
            mx = fmaxf(mx, __shfl_xor(mx, 8, 32));
            const float mn = fmaxf(mrun[r], mx);
            p0[r] = __expf(v0 - mn);
            p1[r] = __expf(v1 - mn);
            float ps = p0[r] + p1[r];
            ps += __shfl_xor(ps, 1, 32);
            ps += __shfl_xor(ps, 2, 32);
            ps += __shfl_xor(ps, 4, 32);
            ps += __shfl_xor(ps, 8, 32);
            alpha[r] = __expf(mrun[r] - mn);
            lrun[r] = lrun[r] * alpha[r] + ps;
            mrun[r] = mn;
        }
#pragma unroll
        for (int r = 0; r < 8; ++r) {
            acc0[r] *= alpha[r]; acc1[r] *= alpha[r];
            acc2[r] *= alpha[r]; acc3[r] *= alpha[r];
        }
        // bounce P (C-layout) through LDS into A-fragment layout
#pragma unroll
        for (int r = 0; r < 8; ++r) {
            const int qrow = r + 8 * hi;
            Pl[qrow * 32 + ln15]      = (_Float16)p0[r];
            Pl[qrow * 32 + 16 + ln15] = (_Float16)p1[r];
        }
        __syncthreads();
        v16h ap = *(const v16h*)(Pl + ln15 * 32 + hi * 16);
        __syncthreads();
        v16h bv0 = *(const v16h*)(Vb + (size_t)(0 + ln15)  * Lkpad + j + hi * 16);
        v16h bv1 = *(const v16h*)(Vb + (size_t)(16 + ln15) * Lkpad + j + hi * 16);
        v16h bv2 = *(const v16h*)(Vb + (size_t)(32 + ln15) * Lkpad + j + hi * 16);
        v16h bv3 = *(const v16h*)(Vb + (size_t)(48 + ln15) * Lkpad + j + hi * 16);
        acc0 = __builtin_amdgcn_wmma_f32_16x16x32_f16(false, ap, false, bv0, (short)0, acc0, false, false);
        acc1 = __builtin_amdgcn_wmma_f32_16x16x32_f16(false, ap, false, bv1, (short)0, acc1, false, false);
        acc2 = __builtin_amdgcn_wmma_f32_16x16x32_f16(false, ap, false, bv2, (short)0, acc2, false, false);
        acc3 = __builtin_amdgcn_wmma_f32_16x16x32_f16(false, ap, false, bv3, (short)0, acc3, false, false);
    }
#pragma unroll
    for (int r = 0; r < 8; ++r) {
        const int qi = q0 + r + 8 * hi;
        if (qi < Lq) {
            const float inv = 1.f / lrun[r];
            const size_t base = ((size_t)b * Lq + qi) * DM + h * EH + ln15;
            Om[base +  0] = acc0[r] * inv;
            Om[base + 16] = acc1[r] * inv;
            Om[base + 32] = acc2[r] * inv;
            Om[base + 48] = acc3[r] * inv;
        }
    }
}

// ================= elementwise / reduction kernels =================
__global__ void decomp_kernel(const float* __restrict__ x, float* __restrict__ seas,
                              float* __restrict__ trend, int L)
{
    long long idx = (long long)blockIdx.x * blockDim.x + threadIdx.x;
    long long n = (long long)NB * L * CIN;
    if (idx >= n) return;
    int c = (int)(idx % CIN);
    int l = (int)((idx / CIN) % L);
    int b = (int)(idx / ((long long)CIN * L));
    float s = 0.f;
    for (int i = -12; i <= 12; ++i) {
        int t = l + i; t = t < 0 ? 0 : (t >= L ? L - 1 : t);
        s += x[((size_t)b * L + t) * CIN + c];
    }
    const float mm = s * (1.f / 25.f);
    seas[idx] = x[idx] - mm;
    if (trend) trend[idx] = mm;
}

__global__ void build_sd_kernel(const float* __restrict__ sdec, float* __restrict__ sd)
{
    long long idx = (long long)blockIdx.x * blockDim.x + threadIdx.x;
    long long n = (long long)NB * LDEC * CIN;
    if (idx >= n) return;
    int c = (int)(idx % CIN);
    int l = (int)((idx / CIN) % LDEC);
    int b = (int)(idx / ((long long)CIN * LDEC));
    sd[idx] = (l < LBL) ? sdec[((size_t)b * LDEC + (LDEC - LBL + l)) * CIN + c] : 0.f;
}

__global__ void im2col3_kernel(const float* __restrict__ x, _Float16* __restrict__ dst, int L)
{
    long long idx = (long long)blockIdx.x * blockDim.x + threadIdx.x;
    long long n = (long long)NB * L * K3;
    if (idx >= n) return;
    int col = (int)(idx % K3);
    long long rl = idx / K3;
    int l = (int)(rl % L);
    int b = (int)(rl / L);
    _Float16 v = (_Float16)0.f;
    if (col < 3 * CIN) {
        int c = col / 3, t = col % 3;
        int lt = l + t - 1; lt = (lt + L) % L;          // circular pad
        v = (_Float16)x[((size_t)b * L + lt) * CIN + c];
    }
    dst[idx] = v;
}

__global__ void add_pos_kernel(float* __restrict__ x, int L)
{
    long long idx = (long long)blockIdx.x * blockDim.x + threadIdx.x;
    long long n = (long long)NB * L * DM;
    if (idx >= n) return;
    int d = (int)(idx % DM);
    int l = (int)((idx / DM) % L);
    const float fi = (float)((d >> 1) * 2);
    const float dv = __expf(fi * (-logf(10000.f) / (float)DM));
    const float ang = (float)l * dv;
    x[idx] += (d & 1) ? cosf(ang) : sinf(ang);
}

__global__ void cvt_a_kernel(const float* __restrict__ src, _Float16* __restrict__ dst,
                             int M, int K, int Kpad)
{
    long long idx = (long long)blockIdx.x * blockDim.x + threadIdx.x;
    long long n = (long long)M * Kpad;
    if (idx >= n) return;
    int k = (int)(idx % Kpad);
    long long m = idx / Kpad;
    dst[idx] = (k < K) ? (_Float16)src[m * K + k] : (_Float16)0.f;
}

__global__ void cvt_w_kernel(const float* __restrict__ src, _Float16* __restrict__ dst,
                             int N, int K, int Kpad)
{
    long long idx = (long long)blockIdx.x * blockDim.x + threadIdx.x;
    long long n = (long long)N * Kpad;
    if (idx >= n) return;
    int k = (int)(idx % Kpad);
    long long nn = idx / Kpad;
    dst[idx] = (k < K) ? (_Float16)src[nn * K + k] : (_Float16)0.f;
}

__global__ void repack_qk_kernel(const float* __restrict__ src, _Float16* __restrict__ dst,
                                 int L, int Lpad)
{
    long long idx = (long long)blockIdx.x * blockDim.x + threadIdx.x;
    long long n = (long long)NB * NH * Lpad * EH;
    if (idx >= n) return;
    int e = (int)(idx % EH);
    int l = (int)((idx / EH) % Lpad);
    int h = (int)((idx / ((long long)EH * Lpad)) % NH);
    int b = (int)(idx / ((long long)EH * Lpad * NH));
    dst[idx] = (l < L) ? (_Float16)src[((size_t)b * L + l) * DM + h * EH + e] : (_Float16)0.f;
}

__global__ void repack_vt_kernel(const float* __restrict__ src, _Float16* __restrict__ dst,
                                 int L, int Lpad)
{
    long long idx = (long long)blockIdx.x * blockDim.x + threadIdx.x;
    long long n = (long long)NB * NH * EH * Lpad;
    if (idx >= n) return;
    int l = (int)(idx % Lpad);
    int e = (int)((idx / Lpad) % EH);
    int h = (int)((idx / ((long long)Lpad * EH)) % NH);
    int b = (int)(idx / ((long long)Lpad * EH * NH));
    dst[idx] = (l < L) ? (_Float16)src[((size_t)b * L + l) * DM + h * EH + e] : (_Float16)0.f;
}

// out = LN(a (+ b)) * g + be   over D=512, one block per row
__global__ __launch_bounds__(256)
void add_ln_kernel(float* __restrict__ out, const float* __restrict__ a,
                   const float* __restrict__ b, const float* __restrict__ g,
                   const float* __restrict__ be, int M)
{
    const int row = blockIdx.x;
    if (row >= M) return;
    const int tid = threadIdx.x;
    const size_t base = (size_t)row * DM;
    float v0 = a[base + tid];
    float v1 = a[base + tid + 256];
    if (b) { v0 += b[base + tid]; v1 += b[base + tid + 256]; }
    __shared__ float sred[8];
    float s = v0 + v1;
    for (int off = 16; off; off >>= 1) s += __shfl_xor(s, off, 32);
    if ((tid & 31) == 0) sred[tid >> 5] = s;
    __syncthreads();
    float tot = 0.f;
    for (int i = 0; i < 8; ++i) tot += sred[i];
    const float mean = tot / (float)DM;
    __syncthreads();
    const float d0 = v0 - mean, d1 = v1 - mean;
    s = d0 * d0 + d1 * d1;
    for (int off = 16; off; off >>= 1) s += __shfl_xor(s, off, 32);
    if ((tid & 31) == 0) sred[tid >> 5] = s;
    __syncthreads();
    tot = 0.f;
    for (int i = 0; i < 8; ++i) tot += sred[i];
    const float rstd = rsqrtf(tot / (float)DM + 1e-5f);
    out[base + tid]       = d0 * rstd * g[tid] + be[tid];
    out[base + tid + 256] = d1 * rstd * g[tid + 256] + be[tid + 256];
}

__global__ void meanstd_kernel(const float* __restrict__ tr, float* __restrict__ mean,
                               float* __restrict__ stdv)
{
    int i = blockIdx.x * blockDim.x + threadIdx.x;
    if (i >= NB * CIN) return;
    int c = i % CIN, b = i / CIN;
    float s = 0.f, s2 = 0.f;
    for (int t = 0; t < SEQ; ++t) {
        float v = tr[((size_t)b * SEQ + t) * CIN + c];
        s += v; s2 += v * v;
    }
    const float m = s / (float)SEQ;
    mean[i] = m;
    stdv[i] = sqrtf(s2 / (float)SEQ - m * m + 1e-5f);
}

__global__ void revin_t_kernel(const float* __restrict__ tr, const float* __restrict__ mean,
                               const float* __restrict__ stdv, const float* __restrict__ rw,
                               const float* __restrict__ rb, _Float16* __restrict__ dst)
{
    long long idx = (long long)blockIdx.x * blockDim.x + threadIdx.x;
    long long n = (long long)NB * CIN * KTR;
    if (idx >= n) return;
    int t = (int)(idx % KTR);
    int c = (int)((idx / KTR) % CIN);
    int b = (int)(idx / ((long long)KTR * CIN));
    _Float16 v = (_Float16)0.f;
    if (t < SEQ) {
        int bc = b * CIN + c;
        v = (_Float16)(((tr[((size_t)b * SEQ + t) * CIN + c] - mean[bc]) / stdv[bc]) * rw[c] + rb[c]);
    }
    dst[idx] = v;
}

__global__ void ratio_kernel(const float* __restrict__ senc, const float* __restrict__ proj,
                             float* __restrict__ ratio)
{
    int i = blockIdx.x * blockDim.x + threadIdx.x;
    if (i >= NB * CIN) return;
    int c = i % CIN, b = i / CIN;
    float sa = 0.f, so = 0.f;
    for (int t = 0; t < SEQ; ++t)  sa += fabsf(senc[((size_t)b * SEQ + t) * CIN + c]);
    for (int t = 0; t < PRED; ++t) so += fabsf(proj[((size_t)b * LDEC + LBL + t) * COUT + c]);
    ratio[i] = (sa / (float)SEQ) / (so / (float)PRED);
}

__global__ void combine_kernel(const float* __restrict__ t3, const float* __restrict__ proj,
                               const float* __restrict__ mean, const float* __restrict__ stdv,
                               const float* __restrict__ rw, const float* __restrict__ rb,
                               const float* __restrict__ ratio, float* __restrict__ out)
{
    long long idx = (long long)blockIdx.x * blockDim.x + threadIdx.x;
    long long n = (long long)NB * PRED * COUT;
    if (idx >= n) return;
    int c = (int)(idx % COUT);
    int t = (int)((idx / COUT) % PRED);
    int b = (int)(idx / ((long long)COUT * PRED));
    int bc = b * CIN + c;
    const float tv = t3[((size_t)b * CIN + c) * PRED + t];
    const float trend = (tv - rb[c]) / (rw[c] + 1e-10f) * stdv[bc] + mean[bc];
    const float so = proj[((size_t)b * LDEC + LBL + t) * COUT + c];
    out[idx] = trend + ratio[bc] * so;
}

// ================= host orchestration =================
static inline long long cdiv(long long a, long long b) { return (a + b - 1) / b; }

extern "C" void kernel_launch(void* const* d_in, const int* in_sizes, int n_in,
                              void* d_out, int out_size, void* d_ws, size_t ws_size,
                              hipStream_t stream)
{
    (void)in_sizes; (void)n_in; (void)out_size; (void)ws_size;
    auto F = [&](int i) { return (const float*)d_in[i]; };

    const float* tau   = F(0);
    const float* delta = F(1);
    const float* x_enc = F(2);
    const float* x_dec = F(4);
    // params, JAX pytree (alphabetically sorted dict) flatten order starting at 6:
    // 6 dec_conv | dec_layer: 7 b1, 8 b2, cross{9 bk,10 bo,11 bq,12 bv,13 wk,14 wo,15 wq,16 wv},
    // 17 n1b,18 n1g,19 n2b,20 n2g,21 n3b,22 n3g, self{23..30}, 31 w1, 32 w2 |
    // 33 dec_nb, 34 dec_ng, 35 enc_conv | enc_layers[0]: attn{36 bk,37 bo,38 bq,39 bv,40 wk,41 wo,
    // 42 wq,43 wv},44 b1,45 b2,46 n1b,47 n1g,48 n2b,49 n2g,50 w1,51 w2 | layer1 = +16 |
    // 68 enc_nb,69 enc_ng,70 proj_b,71 proj_w,72 rev_b,73 rev_w,74 tr_b1,75 tr_b2,76 tr_b3,
    // 77 tr_w1,78 tr_w2,79 tr_w3

    const int M_enc = NB * SEQ, M_dec = NB * LDEC, M_tr = NB * CIN;

    char* ws = (char*)d_ws;
    size_t off = 0;
    auto allocF = [&](size_t n) { float* p = (float*)(ws + off); off += ((n * 4 + 255) / 256) * 256; return p; };
    auto allocH = [&](size_t n) { _Float16* p = (_Float16*)(ws + off); off += ((n * 2 + 255) / 256) * 256; return p; };

    float* s_enc   = allocF((size_t)NB * SEQ * CIN);
    float* t_enc   = allocF((size_t)NB * SEQ * CIN);
    float* s_dec   = allocF((size_t)NB * LDEC * CIN);
    float* sd      = allocF((size_t)NB * LDEC * CIN);
    float* actA    = allocF((size_t)M_dec * DM);
    float* actB    = allocF((size_t)M_dec * DM);
    float* actC    = allocF((size_t)M_dec * DM);
    float* actQ    = allocF((size_t)M_dec * DM);
    float* actK    = allocF((size_t)M_dec * DM);
    float* actV    = allocF((size_t)M_dec * DM);
    float* ffw     = allocF((size_t)M_dec * DFF);
    float* enc_out = allocF((size_t)M_enc * DM);
    float* proj_o  = allocF((size_t)M_dec * COUT);
    float* meanv   = allocF((size_t)M_tr);
    float* stdv    = allocF((size_t)M_tr);
    float* ratio   = allocF((size_t)M_tr);
    float* t1      = allocF((size_t)M_tr * DM);
    float* t2      = allocF((size_t)M_tr * DM);
    float* t3      = allocF((size_t)M_tr * PRED);
    _Float16* a16  = allocH((size_t)M_dec * DFF);
    _Float16* w16  = allocH((size_t)DFF * DM);
    _Float16* q16  = allocH((size_t)NB * NH * LQD * EH);
    _Float16* k16  = allocH((size_t)NB * NH * LKD * EH);
    _Float16* vt16 = allocH((size_t)NB * NH * LKD * EH);
    _Float16* ck16 = allocH((size_t)NB * NH * LKE * EH);
    _Float16* cv16 = allocH((size_t)NB * NH * LKE * EH);

    const float scale = 1.f / sqrtf((float)EH);

    auto gemm = [&](const _Float16* A, int lda, const _Float16* W, int ldw, float* C, int ldc,
                    const float* bias, int M, int N, int K, int relu) {
        dim3 g((unsigned)cdiv(M, 128), (unsigned)cdiv(N, 128));
        gemm_f16_kernel<<<g, 256, 0, stream>>>(A, lda, W, ldw, C, ldc, bias, M, N, K, relu);
    };
    auto cvtA = [&](const float* s, _Float16* d, int M, int K, int Kpad) {
        long long n = (long long)M * Kpad;
        cvt_a_kernel<<<(unsigned)cdiv(n, 256), 256, 0, stream>>>(s, d, M, K, Kpad);
    };
    auto cvtW = [&](const float* s, _Float16* d, int N, int K, int Kpad) {
        long long n = (long long)N * Kpad;
        cvt_w_kernel<<<(unsigned)cdiv(n, 256), 256, 0, stream>>>(s, d, N, K, Kpad);
    };
    auto rep_qk = [&](const float* s, _Float16* d, int L, int Lpad) {
        long long n = (long long)NB * NH * Lpad * EH;
        repack_qk_kernel<<<(unsigned)cdiv(n, 256), 256, 0, stream>>>(s, d, L, Lpad);
    };
    auto rep_vt = [&](const float* s, _Float16* d, int L, int Lpad) {
        long long n = (long long)NB * NH * EH * Lpad;
        repack_vt_kernel<<<(unsigned)cdiv(n, 256), 256, 0, stream>>>(s, d, L, Lpad);
    };
    auto attn = [&](const _Float16* Q, const _Float16* K, const _Float16* V, float* O,
                    const float* dl, int Lq, int Lk, int Lqp, int Lkp, int causal) {
        dim3 g((unsigned)(Lqp / 16), (unsigned)(NB * NH));
        attn_kernel<<<g, 32, 0, stream>>>(Q, K, V, O, tau, dl, Lq, Lk, Lqp, Lkp, causal, scale);
    };
    auto aln = [&](float* o, const float* a, const float* b, const float* g, const float* be, int M) {
        add_ln_kernel<<<(unsigned)M, 256, 0, stream>>>(o, a, b, g, be, M);
    };

    // ---- decompositions ----
    {
        long long n = (long long)NB * SEQ * CIN;
        decomp_kernel<<<(unsigned)cdiv(n, 256), 256, 0, stream>>>(x_enc, s_enc, t_enc, SEQ);
        long long m = (long long)NB * LDEC * CIN;
        decomp_kernel<<<(unsigned)cdiv(m, 256), 256, 0, stream>>>(x_dec, s_dec, nullptr, LDEC);
        build_sd_kernel<<<(unsigned)cdiv(m, 256), 256, 0, stream>>>(s_dec, sd);
    }
    // ---- encoder embedding (im2col + WMMA GEMM + pos) ----
    {
        long long n = (long long)NB * SEQ * K3;
        im2col3_kernel<<<(unsigned)cdiv(n, 256), 256, 0, stream>>>(s_enc, a16, SEQ);
        cvtW(F(35), w16, DM, 3 * CIN, K3);
        gemm(a16, K3, w16, K3, actA, DM, nullptr, M_enc, DM, K3, 0);
        long long p = (long long)NB * SEQ * DM;
        add_pos_kernel<<<(unsigned)cdiv(p, 256), 256, 0, stream>>>(actA, SEQ);
    }
    // ---- encoder layers ----
    for (int li = 0; li < 2; ++li) {
        const int o = 36 + li * 16;
        const float *wq = F(o + 6), *bq = F(o + 2), *wk = F(o + 4), *bk = F(o + 0);
        const float *wv = F(o + 7), *bv = F(o + 3), *wo = F(o + 5), *bo = F(o + 1);
        const float *w1 = F(o + 14), *b1 = F(o + 8), *w2 = F(o + 15), *b2 = F(o + 9);
        const float *n1g = F(o + 11), *n1b = F(o + 10), *n2g = F(o + 13), *n2b = F(o + 12);
        cvtA(actA, a16, M_enc, DM, DM);
        cvtW(wq, w16, DM, DM, DM); gemm(a16, DM, w16, DM, actQ, DM, bq, M_enc, DM, DM, 0);
        cvtW(wk, w16, DM, DM, DM); gemm(a16, DM, w16, DM, actK, DM, bk, M_enc, DM, DM, 0);
        cvtW(wv, w16, DM, DM, DM); gemm(a16, DM, w16, DM, actV, DM, bv, M_enc, DM, DM, 0);
        rep_qk(actQ, q16, SEQ, LQE);
        rep_qk(actK, k16, SEQ, LKE);
        rep_vt(actV, vt16, SEQ, LKE);
        attn(q16, k16, vt16, actB, delta, SEQ, SEQ, LQE, LKE, 0);
        cvtA(actB, a16, M_enc, DM, DM);
        cvtW(wo, w16, DM, DM, DM); gemm(a16, DM, w16, DM, actC, DM, bo, M_enc, DM, DM, 0);
        aln(actB, actA, actC, n1g, n1b, M_enc);
        cvtA(actB, a16, M_enc, DM, DM);
        cvtW(w1, w16, DFF, DM, DM); gemm(a16, DM, w16, DM, ffw, DFF, b1, M_enc, DFF, DM, 1);
        cvtA(ffw, a16, M_enc, DFF, DFF);
        cvtW(w2, w16, DM, DFF, DFF); gemm(a16, DFF, w16, DFF, actC, DM, b2, M_enc, DM, DFF, 0);
        aln(actA, actB, actC, n2g, n2b, M_enc);
    }
    aln(enc_out, actA, nullptr, F(69), F(68), M_enc);   // final encoder LN
    // ---- cross-attention K/V from encoder output ----
    cvtA(enc_out, a16, M_enc, DM, DM);
    cvtW(F(13), w16, DM, DM, DM); gemm(a16, DM, w16, DM, actK, DM, F(9),  M_enc, DM, DM, 0);
    cvtW(F(16), w16, DM, DM, DM); gemm(a16, DM, w16, DM, actV, DM, F(12), M_enc, DM, DM, 0);
    rep_qk(actK, ck16, SEQ, LKE);
    rep_vt(actV, cv16, SEQ, LKE);
    // ---- decoder embedding ----
    {
        long long n = (long long)NB * LDEC * K3;
        im2col3_kernel<<<(unsigned)cdiv(n, 256), 256, 0, stream>>>(sd, a16, LDEC);
        cvtW(F(6), w16, DM, 3 * CIN, K3);
        gemm(a16, K3, w16, K3, actA, DM, nullptr, M_dec, DM, K3, 0);
        long long p = (long long)NB * LDEC * DM;
        add_pos_kernel<<<(unsigned)cdiv(p, 256), 256, 0, stream>>>(actA, LDEC);
    }
    // ---- decoder self-attention (causal) ----
    cvtA(actA, a16, M_dec, DM, DM);
    cvtW(F(29), w16, DM, DM, DM); gemm(a16, DM, w16, DM, actQ, DM, F(25), M_dec, DM, DM, 0);
    cvtW(F(27), w16, DM, DM, DM); gemm(a16, DM, w16, DM, actK, DM, F(23), M_dec, DM, DM, 0);
    cvtW(F(30), w16, DM, DM, DM); gemm(a16, DM, w16, DM, actV, DM, F(26), M_dec, DM, DM, 0);
    rep_qk(actQ, q16, LDEC, LQD);
    rep_qk(actK, k16, LDEC, LKD);
    rep_vt(actV, vt16, LDEC, LKD);
    attn(q16, k16, vt16, actB, nullptr, LDEC, LDEC, LQD, LKD, 1);
    cvtA(actB, a16, M_dec, DM, DM);
    cvtW(F(28), w16, DM, DM, DM); gemm(a16, DM, w16, DM, actC, DM, F(24), M_dec, DM, DM, 0);
    aln(actB, actA, actC, F(18), F(17), M_dec);
    // ---- decoder cross-attention ----
    cvtA(actB, a16, M_dec, DM, DM);
    cvtW(F(15), w16, DM, DM, DM); gemm(a16, DM, w16, DM, actQ, DM, F(11), M_dec, DM, DM, 0);
    rep_qk(actQ, q16, LDEC, LQD);
    attn(q16, ck16, cv16, actC, nullptr, LDEC, SEQ, LQD, LKE, 0);
    cvtA(actC, a16, M_dec, DM, DM);
    cvtW(F(14), w16, DM, DM, DM); gemm(a16, DM, w16, DM, actA, DM, F(10), M_dec, DM, DM, 0);
    aln(actC, actB, actA, F(20), F(19), M_dec);
    // ---- decoder FFN + final LNs ----
    cvtA(actC, a16, M_dec, DM, DM);
    cvtW(F(31), w16, DFF, DM, DM); gemm(a16, DM, w16, DM, ffw, DFF, F(7), M_dec, DFF, DM, 1);
    cvtA(ffw, a16, M_dec, DFF, DFF);
    cvtW(F(32), w16, DM, DFF, DFF); gemm(a16, DFF, w16, DFF, actA, DM, F(8), M_dec, DM, DFF, 0);
    aln(actB, actC, actA, F(22), F(21), M_dec);
    aln(actA, actB, nullptr, F(34), F(33), M_dec);
    // ---- output projection ----
    cvtA(actA, a16, M_dec, DM, DM);
    cvtW(F(71), w16, COUT, DM, DM);
    gemm(a16, DM, w16, DM, proj_o, COUT, F(70), M_dec, COUT, DM, 0);
    // ---- ratio ----
    ratio_kernel<<<(unsigned)cdiv(NB * CIN, 256), 256, 0, stream>>>(s_enc, proj_o, ratio);
    // ---- trend path (RevIN + MLP over time) ----
    meanstd_kernel<<<(unsigned)cdiv(NB * CIN, 256), 256, 0, stream>>>(t_enc, meanv, stdv);
    {
        long long n = (long long)NB * CIN * KTR;
        revin_t_kernel<<<(unsigned)cdiv(n, 256), 256, 0, stream>>>(t_enc, meanv, stdv, F(73), F(72), a16);
    }
    cvtW(F(77), w16, DM, SEQ, KTR); gemm(a16, KTR, w16, KTR, t1, DM, F(74), M_tr, DM, KTR, 1);
    cvtA(t1, a16, M_tr, DM, DM);
    cvtW(F(78), w16, DM, DM, DM);   gemm(a16, DM, w16, DM, t2, DM, F(75), M_tr, DM, DM, 1);
    cvtA(t2, a16, M_tr, DM, DM);
    cvtW(F(79), w16, PRED, DM, DM); gemm(a16, DM, w16, DM, t3, PRED, F(76), M_tr, PRED, DM, 0);
    // ---- combine ----
    {
        long long n = (long long)NB * PRED * COUT;
        combine_kernel<<<(unsigned)cdiv(n, 256), 256, 0, stream>>>(t3, proj_o, meanv, stdv,
                                                                   F(73), F(72), ratio, (float*)d_out);
    }
}
